// Sinkhorn_seq_7859790152169
// MI455X (gfx1250) — compile-verified, matrix operand order checked by reading, not run on hardware
//
#include <hip/hip_runtime.h>
#include <math.h>

// Problem constants from the reference
#define B_ 4
#define S_ 256
#define V_ 32000

// ---------------------------------------------------------------------------
// CDNA5 async global->LDS DMA (ASYNCcnt-tracked). Inline asm so it bypasses
// builtin-signature differences between toolchains. Host pass / non-AMDGCN
// gets a plain copy (bodies are parsed but never codegen'd on host).
// ---------------------------------------------------------------------------
#if defined(__AMDGCN__)
#define ASYNC_CP16(ldsptr, gptr)                                        \
  asm volatile("global_load_async_to_lds_b128 %0, %1, off"              \
               :: "v"((unsigned)(size_t)(ldsptr)),                      \
                  "v"((const void*)(gptr))                              \
               : "memory")
#define WAIT_ASYNC(n) asm volatile("s_wait_asynccnt " #n ::: "memory")
#else
#define ASYNC_CP16(ldsptr, gptr) (*(float4*)(ldsptr) = *(const float4*)(gptr))
#define WAIT_ASYNC(n) do {} while (0)
#endif

typedef __attribute__((ext_vector_type(16))) _Float16 v16h;
typedef __attribute__((ext_vector_type(8)))  float    v8f;

// ---------------------------------------------------------------------------
// K1: softmax(y * 0.5) per row of 32000.  grid = (B*S, 2), block = 256.
// blockIdx.y selects (y_s -> p_s) vs (y_t -> p_t).
// ---------------------------------------------------------------------------
__global__ void __launch_bounds__(256) softmax_rows_kernel(
    const float* __restrict__ ys, const float* __restrict__ yt,
    float* __restrict__ ps, float* __restrict__ pt) {
  const float* y = (blockIdx.y == 0) ? ys : yt;
  float*       p = (blockIdx.y == 0) ? ps : pt;
  const size_t base = (size_t)blockIdx.x * V_;
  const int tid = threadIdx.x;
  __shared__ float red[256];

  float m = -INFINITY;
  for (int i = tid; i < V_; i += 256) m = fmaxf(m, y[base + i]);
  red[tid] = m;
  __syncthreads();
  for (int s = 128; s > 0; s >>= 1) {
    if (tid < s) red[tid] = fmaxf(red[tid], red[tid + s]);
    __syncthreads();
  }
  m = red[0] * 0.5f;   // max(y/2) == max(y)/2
  __syncthreads();

  float sum = 0.0f;
  for (int i = tid; i < V_; i += 256) sum += __expf(y[base + i] * 0.5f - m);
  red[tid] = sum;
  __syncthreads();
  for (int s = 128; s > 0; s >>= 1) {
    if (tid < s) red[tid] += red[tid + s];
    __syncthreads();
  }
  const float inv = 1.0f / red[0];
  __syncthreads();
  for (int i = tid; i < V_; i += 256) p[base + i] = __expf(y[base + i] * 0.5f - m) * inv;
}

// ---------------------------------------------------------------------------
// K2: W[b][i][j] = sum_v |p_s[b,i,v] - p_t[b,j,v]|
// grid = (S/64, S/64, B), block = 256.  64x64 output tile, K-chunks of 32,
// double-buffered async DMA into LDS, 4x4 register micro-tile per thread.
// LDS rows padded to 36 floats (144B, keeps 16B alignment, spreads banks).
// ---------------------------------------------------------------------------
#define TKC   32
#define PITCH 36

__global__ void __launch_bounds__(256) cdist_l1_kernel(
    const float* __restrict__ Ps, const float* __restrict__ Pt,
    float* __restrict__ W) {
  __shared__ float As[2][64 * PITCH];
  __shared__ float Bs[2][64 * PITCH];

  const int tid   = threadIdx.x;
  const int b     = blockIdx.z;
  const int iBase = blockIdx.y * 64;
  const int jBase = blockIdx.x * 64;

  const float* Ag = Ps + (size_t)(b * S_ + iBase) * V_;
  const float* Bg = Pt + (size_t)(b * S_ + jBase) * V_;

  // Each thread moves two 16B chunks per tile (64*32 floats = 512 float4).
  const int t0 = tid,       r0 = t0 >> 3, c0 = (t0 & 7) << 2;
  const int t1 = tid + 256, r1 = t1 >> 3, c1 = (t1 & 7) << 2;

  auto issue = [&](int buf, int k0) {
    ASYNC_CP16(&As[buf][r0 * PITCH + c0], Ag + (size_t)r0 * V_ + k0 + c0);
    ASYNC_CP16(&As[buf][r1 * PITCH + c1], Ag + (size_t)r1 * V_ + k0 + c1);
    ASYNC_CP16(&Bs[buf][r0 * PITCH + c0], Bg + (size_t)r0 * V_ + k0 + c0);
    ASYNC_CP16(&Bs[buf][r1 * PITCH + c1], Bg + (size_t)r1 * V_ + k0 + c1);
  };

  const int tx = tid & 15, ty = tid >> 4;
  float acc[4][4];
#pragma unroll
  for (int e = 0; e < 4; ++e)
#pragma unroll
    for (int f = 0; f < 4; ++f) acc[e][f] = 0.0f;

  const int NK = V_ / TKC;  // 1000
  issue(0, 0);
  for (int kc = 0; kc < NK; ++kc) {
    const int cur = kc & 1;
    if (kc + 1 < NK) {
      issue(cur ^ 1, (kc + 1) * TKC);  // 4 new ops in flight for next buffer
      WAIT_ASYNC(4);                   // current buffer's 4 ops complete
    } else {
      WAIT_ASYNC(0);
    }
    __syncthreads();

#pragma unroll
    for (int k = 0; k < TKC; k += 4) {
      float4 av[4], bv[4];
#pragma unroll
      for (int e = 0; e < 4; ++e)
        av[e] = *(const float4*)&As[cur][(4 * ty + e) * PITCH + k];
#pragma unroll
      for (int e = 0; e < 4; ++e)
        bv[e] = *(const float4*)&Bs[cur][(4 * tx + e) * PITCH + k];
#pragma unroll
      for (int e = 0; e < 4; ++e)
#pragma unroll
        for (int f = 0; f < 4; ++f) {
          acc[e][f] += fabsf(av[e].x - bv[f].x) + fabsf(av[e].y - bv[f].y) +
                       fabsf(av[e].z - bv[f].z) + fabsf(av[e].w - bv[f].w);
        }
    }
    __syncthreads();  // protect buffer (cur) before it is refilled next iter
  }

  float* Wb = W + ((size_t)b << 16);
#pragma unroll
  for (int e = 0; e < 4; ++e)
#pragma unroll
    for (int f = 0; f < 4; ++f)
      Wb[(size_t)(iBase + 4 * ty + e) * S_ + (jBase + 4 * tx + f)] = acc[e][f];
}

// ---------------------------------------------------------------------------
// K3: Sinkhorn. grid = B, block = 1024 (32 waves). Whole P (256KB) lives in
// LDS (320KB/WGP on CDNA5). Row/col sums done with V_WMMA_F32_16X16X32_F16
// against a ones matrix: D[m,n] = sum_k A[m,k]  (B=ones) or sum_k B[k,n]
// (A=ones), f32 accumulate. Waves 0..15 each own a 16-row / 16-col stripe.
// ---------------------------------------------------------------------------
__global__ void __launch_bounds__(1024) sinkhorn_kernel(
    const float* __restrict__ W, float* __restrict__ out) {
  extern __shared__ float smem[];
  float* P    = smem;                 // 65536 floats
  float* rsum = smem + S_ * S_;       // 256
  float* csum = rsum + S_;            // 256
  float* redw = csum + S_;            // 32

  const int tid  = threadIdx.x;
  const int lane = tid & 31;
  const int wave = tid >> 5;
  const int half = lane >> 4;         // 0: lanes 0-15, 1: lanes 16-31
  const int sub  = lane & 15;
  const float* Wb = W + ((size_t)blockIdx.x << 16);

  for (int i = tid; i < S_ * S_; i += 1024) P[i] = __expf(-10.0f * Wb[i]);
  __syncthreads();

  v16h ones;
#pragma unroll
  for (int i = 0; i < 16; ++i) ones[i] = (_Float16)1.0f;

  for (int it = 0; it < 10; ++it) {
    // ---- row sums: D = A(P-stripe) x ones, every column of D == rowsum ----
    if (wave < 16) {
      const int row = 16 * wave + sub;
      v8f acc = {};
#pragma unroll
      for (int c = 0; c < 8; ++c) {
        const int kb = 32 * c;
        v16h a;
        // 16-bit A layout (16x32): lanes 0-15 hold K in {0..7,16..23},
        // lanes 16-31 hold K in {8..15,24..31}; VGPR v packs (k, k+1).
#pragma unroll
        for (int v = 0; v < 8; ++v) {
          const int k = ((v < 4) ? (2 * v) : (16 + 2 * (v - 4))) + (half ? 8 : 0);
          const float2 pv = *(const float2*)&P[row * S_ + kb + k];
          a[2 * v]     = (_Float16)pv.x;
          a[2 * v + 1] = (_Float16)pv.y;
        }
        acc = __builtin_amdgcn_wmma_f32_16x16x32_f16(false, a, false, ones,
                                                     (short)0, acc, false, false);
      }
      // C/D layout: VGPR r = row M=r (lanes 0-15) / M=r+8 (lanes 16-31)
      if (sub == 0) {
#pragma unroll
        for (int r = 0; r < 8; ++r) rsum[16 * wave + 8 * half + r] = acc[r];
      }
    }
    __syncthreads();
    if (tid < S_) rsum[tid] = 1.0f / rsum[tid];
    __syncthreads();
    for (int i = tid; i < S_ * S_; i += 1024) P[i] *= rsum[i >> 8];
    __syncthreads();

    // ---- col sums: D = ones x B(P-stripe), every row of D == colsum ----
    if (wave < 16) {
      const int col = 16 * wave + sub;
      v8f acc = {};
#pragma unroll
      for (int c = 0; c < 8; ++c) {
        const int kb = 32 * c;
        v16h bm;
        // B (32x16), contiguous k order (A=ones): lanes 0-15 hold K=0..15, lanes 16-31 hold
        // K=16..31; VGPR v packs (k, k+1). Lane index = column N.
#pragma unroll
        for (int v = 0; v < 8; ++v) {
          const int k = (half ? 16 : 0) + 2 * v;
          bm[2 * v]     = (_Float16)P[(kb + k) * S_ + col];
          bm[2 * v + 1] = (_Float16)P[(kb + k + 1) * S_ + col];
        }
        acc = __builtin_amdgcn_wmma_f32_16x16x32_f16(false, ones, false, bm,
                                                     (short)0, acc, false, false);
      }
      if (lane < 16) csum[col] = acc[0];  // D VGPR0, lanes 0-15 = row M=0
    }
    __syncthreads();
    if (tid < S_) csum[tid] = 1.0f / csum[tid];
    __syncthreads();
    for (int i = tid; i < S_ * S_; i += 1024) P[i] *= csum[i & (S_ - 1)];
    __syncthreads();
  }

  // loss_b = sum(P * W); out += 0.001 * loss_b
  float acc = 0.0f;
  for (int i = tid; i < S_ * S_; i += 1024) acc += P[i] * Wb[i];
#pragma unroll
  for (int off = 16; off > 0; off >>= 1) acc += __shfl_xor(acc, off, 32);
  if (lane == 0) redw[wave] = acc;
  __syncthreads();
  if (wave == 0) {
    float v = redw[lane];
#pragma unroll
    for (int off = 16; off > 0; off >>= 1) v += __shfl_xor(v, off, 32);
    if (lane == 0) atomicAdd(out, 0.001f * v);
  }
}

// ---------------------------------------------------------------------------
// Launch: softmax -> cdist -> (zero out) -> sinkhorn, all on `stream`.
// Workspace layout: p_s [B*S*V f32] | p_t [B*S*V f32] | W [B*S*S f32]
// (~263 MB total).
// ---------------------------------------------------------------------------
extern "C" void kernel_launch(void* const* d_in, const int* in_sizes, int n_in,
                              void* d_out, int out_size, void* d_ws, size_t ws_size,
                              hipStream_t stream) {
  (void)in_sizes; (void)n_in; (void)out_size; (void)ws_size;
  const float* y_s = (const float*)d_in[0];
  const float* y_t = (const float*)d_in[1];
  float* out = (float*)d_out;

  float* Ps = (float*)d_ws;
  float* Pt = Ps + (size_t)B_ * S_ * V_;
  float* W  = Pt + (size_t)B_ * S_ * V_;

  softmax_rows_kernel<<<dim3(B_ * S_, 2), 256, 0, stream>>>(y_s, y_t, Ps, Pt);
  cdist_l1_kernel<<<dim3(S_ / 64, S_ / 64, B_), 256, 0, stream>>>(Ps, Pt, W);
  hipMemsetAsync(d_out, 0, sizeof(float), stream);
  const size_t smemBytes = (size_t)(S_ * S_ + S_ + S_ + 32) * sizeof(float);
  sinkhorn_kernel<<<dim3(B_), 1024, smemBytes, stream>>>(W, out);
}
